// Attention_11974368821689
// MI455X (gfx1250) — compile-verified, matrix operand order checked by reading, not run on hardware
//
#include <hip/hip_runtime.h>

typedef __bf16 bf16;
typedef __attribute__((ext_vector_type(8)))  bf16  v8bf;
typedef __attribute__((ext_vector_type(16))) bf16  v16bf;
typedef __attribute__((ext_vector_type(8)))  float v8f;
typedef __attribute__((ext_vector_type(4)))  int   v4i;

#define T_SEQ   4096
#define DMODEL  2048
#define NH      16
#define NKV     8
#define HD      128
#define WINDOW  1024
#define SOFTCAP 50.0f
#define NEG_BIG -3.0e38f
#define Q_SCALE 0.08838834764831845f   // 128^-0.5

// ---------------------------------------------------------------------------
// Async global->LDS staging (CDNA5 GLOBAL_LOAD_ASYNC_TO_LDS_B128, ASYNCcnt).
// Probe-confirmed signature: (v4i* gsrc, v4i* ldst, imm int offset, imm cpol).
// Guarded so the build falls back to vectorized synchronous copies if the
// toolchain lacks the builtins.
// ---------------------------------------------------------------------------
#if defined(__has_builtin)
#if __has_builtin(__builtin_amdgcn_global_load_async_to_lds_b128) && \
    __has_builtin(__builtin_amdgcn_s_wait_asynccnt)
#define HAVE_ASYNC_LDS 1
#endif
#endif

#ifdef HAVE_ASYNC_LDS
static __device__ __forceinline__ void stage16(const bf16* gsrc, bf16* ldst) {
  __builtin_amdgcn_global_load_async_to_lds_b128((v4i*)gsrc, (v4i*)ldst, 0, 0);
}
static __device__ __forceinline__ void stage_wait() {
  __builtin_amdgcn_s_wait_asynccnt(0);
}
#else
static __device__ __forceinline__ void stage16(const bf16* gsrc, bf16* ldst) {
  *(uint4*)ldst = *(const uint4*)gsrc;
}
static __device__ __forceinline__ void stage_wait() {}
#endif

// ---------------------------------------------------------------------------
// WMMA fragment loader (CDNA5 16-bit A/B layout, wave32):
//   lane&15 -> row of the (possibly transposed-source) 16x32 block
//   lane>>4 -> K halves {0..7,16..23} vs {8..15,24..31}
// ---------------------------------------------------------------------------
static __device__ __forceinline__ v16bf load_frag(const bf16* base, int ld, int lane) {
  const int m    = lane & 15;
  const int half = (lane >> 4) & 1;
  const bf16* r  = base + m * ld + 8 * half;
  v8bf lo = *(const v8bf*)(r);
  v8bf hi = *(const v8bf*)(r + 16);
  return __builtin_shufflevector(lo, hi, 0,1,2,3,4,5,6,7,8,9,10,11,12,13,14,15);
}

static __device__ __forceinline__ v8f wmma_bf16(v16bf a, v16bf b, v8f c) {
  return __builtin_amdgcn_wmma_f32_16x16x32_bf16(false, a, false, b, (short)0, c, false, false);
}

// ---------------------------------------------------------------------------
// fp32 -> bf16 (for x)
// ---------------------------------------------------------------------------
__global__ void f32_to_bf16_kernel(const float* __restrict__ in, bf16* __restrict__ out, int n) {
  int i = blockIdx.x * blockDim.x + threadIdx.x;
  if (i < n) out[i] = (bf16)in[i];
}

// ---------------------------------------------------------------------------
// Build TRANSPOSED fused Wqkv^T [4096][2048] bf16 (row = output column).
// row n<2048: Q head n>>7, h=n&127 : Wt[n][d] = q_w[n>>7][d][h]
// rows 2048..3071 -> K, 3072..4095 -> V from kv_w.
// ---------------------------------------------------------------------------
__global__ void build_wqkvT_kernel(const float* __restrict__ qw, const float* __restrict__ kvw,
                                   bf16* __restrict__ wt) {
  int i = blockIdx.x * blockDim.x + threadIdx.x;   // over 4096*2048
  if (i >= 4096 * DMODEL) return;
  int col = i >> 11;           // output column (0..4095)
  int d   = i & 2047;          // k index
  float v;
  if (col < 2048) {
    int n = col >> 7, h = col & 127;
    v = qw[((size_t)n * DMODEL + d) * HD + h];
  } else {
    int which = (col < 3072) ? 0 : 1;
    int c = col - 2048 - which * 1024;
    int kh = c >> 7, h = c & 127;
    v = kvw[(((size_t)which * NKV + kh) * DMODEL + d) * HD + h];
  }
  wt[i] = (bf16)v;
}

// ---------------------------------------------------------------------------
// Build Wo^T [2048][2048]: Wo^T[d][n*128+h] = o_w[n][h][d]  (o_w flat = [r][d])
// ---------------------------------------------------------------------------
__global__ void build_woT_kernel(const float* __restrict__ ow, bf16* __restrict__ wt) {
  int i = blockIdx.x * blockDim.x + threadIdx.x;   // over 2048*2048
  if (i >= DMODEL * DMODEL) return;
  int d = i >> 11;
  int r = i & 2047;
  wt[i] = (bf16)ow[(size_t)r * DMODEL + d];
}

// ---------------------------------------------------------------------------
// bf16 GEMM: C[M,N](fp32) = A[M,K] @ Bt[N,K]^T, A/Bt bf16 row-major.
// 128x128 tile, BK=64, 256 threads (8 waves). Both tiles async-staged to LDS;
// 16 WMMAs per wave per k-step.  Requires M%128==0, N%128==0, K%64==0.
// ---------------------------------------------------------------------------
__global__ void gemm_bf16_kernel(const bf16* __restrict__ A, const bf16* __restrict__ Bt,
                                 float* __restrict__ C, int M, int N, int K) {
  __shared__ bf16 As[128 * 64];   // [m][k]
  __shared__ bf16 Bs[128 * 64];   // [n][k]
  const int tid  = threadIdx.x;
  const int wave = tid >> 5, lane = tid & 31;
  const int m0 = blockIdx.y * 128, n0 = blockIdx.x * 128;
  const int half = (lane >> 4) & 1, nl = lane & 15;

  v8f zero = {};
  v8f acc[8];
#pragma unroll
  for (int i = 0; i < 8; ++i) acc[i] = zero;

  for (int k0 = 0; k0 < K; k0 += 64) {
    // 1024 16B chunks per tile, 4 per thread, per-lane async b128 copies
#pragma unroll
    for (int it = 0; it < 4; ++it) {
      int chunk = tid + it * 256;
      int r = chunk >> 3, c8 = (chunk & 7) * 8;
      stage16(A  + (size_t)(m0 + r) * K + k0 + c8, &As[r * 64 + c8]);
      stage16(Bt + (size_t)(n0 + r) * K + k0 + c8, &Bs[r * 64 + c8]);
    }
    stage_wait();
    __syncthreads();

#pragma unroll
    for (int kc = 0; kc < 2; ++kc) {
      v16bf a = load_frag(&As[(wave * 16) * 64 + kc * 32], 64, lane);
#pragma unroll
      for (int ct = 0; ct < 8; ++ct) {
        v16bf b = load_frag(&Bs[(ct * 16) * 64 + kc * 32], 64, lane);
        acc[ct] = wmma_bf16(a, b, acc[ct]);
      }
    }
    __syncthreads();
  }

#pragma unroll
  for (int ct = 0; ct < 8; ++ct)
#pragma unroll
    for (int i = 0; i < 8; ++i) {
      int row = m0 + wave * 16 + half * 8 + i;
      int col = n0 + ct * 16 + nl;
      C[(size_t)row * N + col] = acc[ct][i];
    }
}

// ---------------------------------------------------------------------------
// RoPE + q-scale + split fused qkv (fp32 [T][4096]) into:
//   Q[16][T][128] bf16 (head-major), K[8][T][128] bf16,
//   V^T[8][128][T] bf16  (so attention stages V rows contiguously)
// ---------------------------------------------------------------------------
__global__ void rope_split_kernel(const float* __restrict__ qkv, bf16* __restrict__ Qo,
                                  bf16* __restrict__ Ko, bf16* __restrict__ VTo) {
  const int t = blockIdx.x;
  const int tid = threadIdx.x;
  const float ft = (float)t;

  for (int idx = tid; idx < NH * 64; idx += blockDim.x) {          // Q heads
    int n = idx >> 6, i = idx & 63;
    const float* base = qkv + (size_t)t * 4096 + n * HD;
    float x1 = base[i], x2 = base[i + 64];
    float ang = ft * __powf(10000.0f, -(float)i * (1.0f / 64.0f));
    float s, c;
    __sincosf(ang, &s, &c);
    bf16* ob = Qo + ((size_t)n * T_SEQ + t) * HD;
    ob[i]      = (bf16)((x1 * c - x2 * s) * Q_SCALE);
    ob[i + 64] = (bf16)((x2 * c + x1 * s) * Q_SCALE);
  }
  for (int idx = tid; idx < NKV * 64; idx += blockDim.x) {         // K heads
    int kh = idx >> 6, i = idx & 63;
    const float* base = qkv + (size_t)t * 4096 + 2048 + kh * HD;
    float x1 = base[i], x2 = base[i + 64];
    float ang = ft * __powf(10000.0f, -(float)i * (1.0f / 64.0f));
    float s, c;
    __sincosf(ang, &s, &c);
    bf16* ob = Ko + ((size_t)kh * T_SEQ + t) * HD;
    ob[i]      = (bf16)(x1 * c - x2 * s);
    ob[i + 64] = (bf16)(x2 * c + x1 * s);
  }
  for (int idx = tid; idx < NKV * HD; idx += blockDim.x) {         // V transposed
    int kh = idx >> 7, h = idx & 127;
    VTo[((size_t)kh * HD + h) * T_SEQ + t] =
        (bf16)qkv[(size_t)t * 4096 + 3072 + kh * HD + h];
  }
}

// ---------------------------------------------------------------------------
// Flash attention (sliding-window causal + tanh soft-cap).
// 64-query block, 128 threads (4 waves).  Q frags in registers, K frags
// streamed from global, V^T tile async-staged to LDS (overlapped with QK^T).
// ---------------------------------------------------------------------------
__global__ void flash_attn_kernel(const bf16* __restrict__ Q, const bf16* __restrict__ Kc,
                                  const bf16* __restrict__ VTg, bf16* __restrict__ enc) {
  __shared__ bf16  Vt[HD * 64];     // Vt[h][s]
  __shared__ float Sf[64 * 64];     // logits tile (fp32)
  __shared__ bf16  Ps[64 * 64];     // probability tile (bf16)
  __shared__ float mrow[64], lrow[64], frow[64];

  const int tid  = threadIdx.x;
  const int wave = tid >> 5, lane = tid & 31;
  const int head = blockIdx.y, kvh = head >> 1;     // 2 query heads per kv head
  const int m0 = blockIdx.x * 64;
  const int half = (lane >> 4) & 1, nl = lane & 15;
  const int mbase = wave * 16;

  // Q fragments for this wave's 16 rows, full head_dim (4 k-chunks of 32)
  v16bf qf[4];
  const bf16* qbase = Q + ((size_t)head * T_SEQ + m0 + mbase) * HD;
#pragma unroll
  for (int kc = 0; kc < 4; ++kc) qf[kc] = load_frag(qbase + kc * 32, HD, lane);

  v8f zero = {};
  v8f oacc[8];
#pragma unroll
  for (int i = 0; i < 8; ++i) oacc[i] = zero;

  if (tid < 64) { mrow[tid] = NEG_BIG; lrow[tid] = 0.0f; }
  __syncthreads();

  int s_lo = m0 - (WINDOW - 1); if (s_lo < 0) s_lo = 0;
  const int s_start = (s_lo >> 6) << 6;
  const int s_end   = m0 + 63;

  for (int s0 = s_start; s0 <= s_end; s0 += 64) {
    // async-stage V^T tile [128][64]; overlapped with the QK^T WMMAs below
#pragma unroll
    for (int it = 0; it < 8; ++it) {
      int chunk = tid + it * 128;
      int h = chunk >> 3, c8 = (chunk & 7) * 8;
      stage16(VTg + ((size_t)kvh * HD + h) * T_SEQ + s0 + c8, &Vt[h * 64 + c8]);
    }
    // prefetch next K tile
    __builtin_prefetch(Kc + ((size_t)kvh * T_SEQ + s0 + 64) * HD + lane * 8, 0, 0);

    // S = Q K^T (16 WMMAs/wave), soft-cap + window/causal mask -> Sf
#pragma unroll
    for (int ct = 0; ct < 4; ++ct) {
      v8f sacc = zero;
#pragma unroll
      for (int kc = 0; kc < 4; ++kc) {
        v16bf kb = load_frag(Kc + ((size_t)kvh * T_SEQ + s0 + ct * 16) * HD + kc * 32, HD, lane);
        sacc = wmma_bf16(qf[kc], kb, sacc);
      }
      const int s_col = s0 + ct * 16 + nl;
#pragma unroll
      for (int i = 0; i < 8; ++i) {
        int mloc = mbase + half * 8 + i;
        int t = m0 + mloc;
        float v = SOFTCAP * tanhf(sacc[i] * (1.0f / SOFTCAP));
        bool valid = (s_col <= t) && ((t - s_col) < WINDOW);
        Sf[mloc * 64 + ct * 16 + nl] = valid ? v : NEG_BIG;
      }
    }
    stage_wait();
    __syncthreads();

    // online softmax: one thread per row
    if (tid < 64) {
      const int r = tid;
      float mold = mrow[r];
      float tmax = NEG_BIG;
      for (int c = 0; c < 64; ++c) tmax = fmaxf(tmax, Sf[r * 64 + c]);
      float mnew = fmaxf(mold, tmax);
      float factor = (mold <= -1e37f) ? 0.0f : __expf(mold - mnew);
      float l = lrow[r] * factor;
      for (int c = 0; c < 64; ++c) {
        float v = Sf[r * 64 + c];
        float p = (v <= -1e37f) ? 0.0f : __expf(v - mnew);
        Ps[r * 64 + c] = (bf16)p;
        l += p;
      }
      mrow[r] = mnew; lrow[r] = l; frow[r] = factor;
    }
    __syncthreads();

    // rescale running O accumulators
    float fsc[8];
#pragma unroll
    for (int i = 0; i < 8; ++i) fsc[i] = frow[mbase + half * 8 + i];
#pragma unroll
    for (int ot = 0; ot < 8; ++ot)
#pragma unroll
      for (int i = 0; i < 8; ++i) oacc[ot][i] *= fsc[i];

    // O += P V  (16 WMMAs/wave)
    v16bf pa[2];
#pragma unroll
    for (int kc = 0; kc < 2; ++kc) pa[kc] = load_frag(&Ps[mbase * 64 + kc * 32], 64, lane);
#pragma unroll
    for (int ot = 0; ot < 8; ++ot) {
#pragma unroll
      for (int kc = 0; kc < 2; ++kc) {
        v16bf vb = load_frag(&Vt[(ot * 16) * 64 + kc * 32], 64, lane);
        oacc[ot] = wmma_bf16(pa[kc], vb, oacc[ot]);
      }
    }
    __syncthreads();
  }

  // normalize + write encoded (bf16, [T][2048])
  float linv[8];
#pragma unroll
  for (int i = 0; i < 8; ++i) {
    float l = lrow[mbase + half * 8 + i];
    linv[i] = (l > 0.0f) ? (1.0f / l) : 0.0f;
  }
#pragma unroll
  for (int ot = 0; ot < 8; ++ot)
#pragma unroll
    for (int i = 0; i < 8; ++i) {
      int mloc = mbase + half * 8 + i;
      enc[(size_t)(m0 + mloc) * DMODEL + head * HD + ot * 16 + nl] =
          (bf16)(oacc[ot][i] * linv[i]);
    }
}

// ---------------------------------------------------------------------------
extern "C" void kernel_launch(void* const* d_in, const int* in_sizes, int n_in,
                              void* d_out, int out_size, void* d_ws, size_t ws_size,
                              hipStream_t stream) {
  const float* x   = (const float*)d_in[0];
  // d_in[1] segment_pos (= arange), d_in[2] attn_mask (causal) -- recomputed analytically
  const float* qw  = (const float*)d_in[3];
  const float* kvw = (const float*)d_in[4];
  const float* ow  = (const float*)d_in[5];
  float* out = (float*)d_out;

  char* ws = (char*)d_ws;
  bf16*  xb    = (bf16*)ws;  ws += (size_t)T_SEQ * DMODEL * 2;
  bf16*  wqkvt = (bf16*)ws;  ws += (size_t)4096 * DMODEL * 2;
  float* qkvf  = (float*)ws; ws += (size_t)T_SEQ * 4096 * 4;
  bf16*  qb    = (bf16*)ws;  ws += (size_t)NH  * T_SEQ * HD * 2;
  bf16*  kb    = (bf16*)ws;  ws += (size_t)NKV * T_SEQ * HD * 2;
  bf16*  vtb   = (bf16*)ws;  ws += (size_t)NKV * HD * T_SEQ * 2;
  bf16*  encb  = (bf16*)ws;  ws += (size_t)T_SEQ * DMODEL * 2;
  bf16*  wot   = (bf16*)ws;  ws += (size_t)DMODEL * DMODEL * 2;

  f32_to_bf16_kernel<<<(T_SEQ * DMODEL + 255) / 256, 256, 0, stream>>>(x, xb, T_SEQ * DMODEL);
  build_wqkvT_kernel<<<(4096 * DMODEL + 255) / 256, 256, 0, stream>>>(qw, kvw, wqkvt);
  gemm_bf16_kernel<<<dim3(4096 / 128, T_SEQ / 128), 256, 0, stream>>>(xb, wqkvt, qkvf,
                                                                      T_SEQ, 4096, DMODEL);
  rope_split_kernel<<<T_SEQ, 256, 0, stream>>>(qkvf, qb, kb, vtb);
  flash_attn_kernel<<<dim3(T_SEQ / 64, NH), 128, 0, stream>>>(qb, kb, vtb, encb);
  build_woT_kernel<<<(DMODEL * DMODEL + 255) / 256, 256, 0, stream>>>(ow, wot);
  gemm_bf16_kernel<<<dim3(DMODEL / 128, T_SEQ / 128), 256, 0, stream>>>(encb, wot, out,
                                                                        T_SEQ, DMODEL, DMODEL);
}
